// SegmentalLM_61125974557451
// MI455X (gfx1250) — matrix operand order repaired and checked by reading, not compiled, not run on hardware
//
#include <hip/hip_runtime.h>
#include <hip/hip_bf16.h>

// ---------------------------------------------------------------------------
// Segmental LM for MI455X (gfx1250): all GEMMs via v_wmma_f32_16x16x32_bf16.
// ---------------------------------------------------------------------------

typedef __bf16 bf16_t;
typedef __attribute__((ext_vector_type(16))) __bf16 v16bf;
typedef __attribute__((ext_vector_type(8)))  float  v8f;

#define LOGINF_F 1.0e6f
#define NEG_BIG  (-3.0e38f)

// Problem constants
#define TT 128
#define BB 16
#define EE 256
#define HH 256
#define VV 8000
#define MM 4
#define JN 126            // T - 2
#define NN (JN * BB)      // 2016
#define QQ ((MM + 1) * NN) // 10080

// ---- WMMA fragment loaders -------------------------------------------------
// A fragment: 16x32 bf16, row-major source, row stride ld (elements).
// lane L: row m = L&15; K-base = (L>=16)?16:0; elems [0..15]=K..K+15
__device__ __forceinline__ v16bf load_a_tile(const bf16_t* p0, int ld, int lane) {
  int m = lane & 15;
  int koff = (lane & 16) ? 16 : 0;
  const bf16_t* p = p0 + (size_t)m * ld + koff;
  v16bf a;
#pragma unroll
  for (int i = 0; i < 8; ++i) a[i] = p[i];
#pragma unroll
  for (int i = 0; i < 8; ++i) a[8 + i] = p[8 + i];
  return a;
}

// B fragment: 32x16 bf16 from pre-transposed Bt[N][K].
// lane L: col n = L&15; K-base = (L>=16)?16:0; 16 contiguous bf16.
__device__ __forceinline__ v16bf load_b_tile(const bf16_t* bt, int ld, int n0, int k0, int lane) {
  int n = lane & 15;
  int koff = (lane & 16) ? 16 : 0;
  const bf16_t* p = bt + (size_t)(n0 + n) * ld + k0 + koff;
  v16bf b;
#pragma unroll
  for (int i = 0; i < 16; ++i) b[i] = p[i];
  return b;
}

__device__ __forceinline__ v8f wmma_bf16(v16bf a, v16bf b, v8f c) {
  return __builtin_amdgcn_wmma_f32_16x16x32_bf16(false, a, false, b, (short)0, c, false, false);
}

__device__ __forceinline__ float sigf(float x) { return 1.0f / (1.0f + __expf(-x)); }

// ---- Prep kernels ----------------------------------------------------------
// src [K][N] f32 -> dst [N][K] bf16
__global__ void k_transpose_bf16(const float* __restrict__ src, bf16_t* __restrict__ dst,
                                 int K, int N) {
  int idx = blockIdx.x * 256 + threadIdx.x;
  if (idx >= K * N) return;
  int k = idx / N, n = idx % N;
  dst[(size_t)n * K + k] = (bf16_t)src[idx];
}

// embeds_bf16[t][b][e] = emb[x[b][t]][e]
__global__ void k_embed(const int* __restrict__ x, const float* __restrict__ emb,
                        bf16_t* __restrict__ embeds) {
  int idx = blockIdx.x * 256 + threadIdx.x;
  if (idx >= TT * BB * EE) return;
  int e = idx & 255;
  int b = (idx >> 8) & 15;
  int t = idx >> 12;
  int tok = x[b * TT + t];
  embeds[idx] = (bf16_t)emb[(size_t)tok * EE + e];
}

// tgt_col[q] for q = mstep*2016 + (j*16+b): token x[b][min(j+1+mstep,127)] (mstep<4), else -1
__global__ void k_tgtcol(const int* __restrict__ x, int* __restrict__ tgt_col) {
  int q = blockIdx.x * 256 + threadIdx.x;
  if (q >= QQ) return;
  int mstep = q / NN, nr = q % NN;
  if (mstep >= MM) { tgt_col[q] = -1; return; }
  int j = nr >> 4, b = nr & 15;
  int t = j + 1 + mstep; if (t > TT - 1) t = TT - 1;
  tgt_col[q] = x[b * TT + t];
}

// ---- Encoder LSTM: 1 block, 16 waves, 128 sequential steps -----------------
__global__ __launch_bounds__(512) void k_enc_lstm(
    const bf16_t* __restrict__ embeds,   // [T][16][256]
    const bf16_t* __restrict__ wih_t,    // [1024][256]
    const bf16_t* __restrict__ whh_t,    // [1024][256]
    const float*  __restrict__ bias,     // [1024]
    bf16_t* __restrict__ enc_out)        // [T][16][256] = 0.5*h  (bf16)
{
  __shared__ bf16_t h_sh[BB * HH];
  const int tid  = threadIdx.x;
  const int lane = tid & 31;
  const int w    = tid >> 5;        // 16 waves
  const int jb   = w * 16;          // H column block owned by this wave
  const int n    = lane & 15;
  const int hi   = (lane >> 4) & 1;

  for (int i = tid; i < BB * HH; i += 512) h_sh[i] = (bf16_t)0.0f;
  __syncthreads();

  float bias_r[4];
#pragma unroll
  for (int g = 0; g < 4; ++g) bias_r[g] = bias[g * HH + jb + n];

  v8f c = {};   // cell state, slot r -> row (r+8*hi), col jb+n

  for (int t = 0; t < TT; ++t) {
    v8f acc[4];
#pragma unroll
    for (int g = 0; g < 4; ++g)
#pragma unroll
      for (int r = 0; r < 8; ++r) acc[g][r] = bias_r[g];

    const bf16_t* xrow = embeds + (size_t)t * BB * EE;
#pragma unroll 2
    for (int kk = 0; kk < 8; ++kk) {
      v16bf a = load_a_tile(xrow + kk * 32, EE, lane);
#pragma unroll
      for (int g = 0; g < 4; ++g) {
        v16bf b = load_b_tile(wih_t, EE, g * HH + jb, kk * 32, lane);
        acc[g] = wmma_bf16(a, b, acc[g]);
      }
    }
#pragma unroll 2
    for (int kk = 0; kk < 8; ++kk) {
      v16bf a = load_a_tile(h_sh + kk * 32, HH, lane);
#pragma unroll
      for (int g = 0; g < 4; ++g) {
        v16bf b = load_b_tile(whh_t, HH, g * HH + jb, kk * 32, lane);
        acc[g] = wmma_bf16(a, b, acc[g]);
      }
    }
    __syncthreads();   // everyone done reading old h
    float hv[8];
#pragma unroll
    for (int r = 0; r < 8; ++r) {
      float ig = sigf(acc[0][r]);
      float fg = sigf(acc[1][r]);
      float gg = tanhf(acc[2][r]);
      float og = sigf(acc[3][r]);
      c[r] = fg * c[r] + ig * gg;
      hv[r] = og * tanhf(c[r]);
    }
#pragma unroll
    for (int r = 0; r < 8; ++r) {
      int m = r + 8 * hi;
      h_sh[m * HH + jb + n] = (bf16_t)hv[r];
      enc_out[(size_t)t * BB * HH + m * HH + jb + n] = (bf16_t)(0.5f * hv[r]);
    }
    __syncthreads();   // new h visible before next step
  }
}

// ---- seg_start = tanh(enc_out @ W_start + b_start) -------------------------
__global__ __launch_bounds__(256) void k_seg_start(
    const bf16_t* __restrict__ enc_out,  // [2048][256]
    const bf16_t* __restrict__ wst_t,    // [256][256]
    const float*  __restrict__ b_start,  // [256]
    bf16_t* __restrict__ seg)            // [2048][256]
{
  int gw   = (blockIdx.x * 256 + threadIdx.x) >> 5;  // 2048 waves total
  int lane = threadIdx.x & 31;
  int rt   = gw >> 4;      // 128 row tiles
  int ct   = gw & 15;      // 16 col tiles
  int n    = lane & 15;
  int hi   = (lane >> 4) & 1;

  v8f acc;
  float bv = b_start[ct * 16 + n];
#pragma unroll
  for (int r = 0; r < 8; ++r) acc[r] = bv;

  const bf16_t* arow = enc_out + (size_t)rt * 16 * HH;
#pragma unroll
  for (int kk = 0; kk < 8; ++kk) {
    v16bf a = load_a_tile(arow + kk * 32, HH, lane);
    v16bf b = load_b_tile(wst_t, HH, ct * 16, kk * 32, lane);
    acc = wmma_bf16(a, b, acc);
  }
#pragma unroll
  for (int r = 0; r < 8; ++r) {
    int row = rt * 16 + r + 8 * hi;
    seg[(size_t)row * HH + ct * 16 + n] = (bf16_t)tanhf(acc[r]);
  }
}

// ---- Decoder LSTM: 126 blocks (row stripes), 4 steps each ------------------
__global__ __launch_bounds__(512) void k_dec_lstm(
    const bf16_t* __restrict__ embeds,   // [128][16][256]
    const bf16_t* __restrict__ seg,      // [2048][256]
    const bf16_t* __restrict__ wih_t,    // [1024][256]
    const bf16_t* __restrict__ whh_t,    // [1024][256]
    const float*  __restrict__ bias,     // [1024]
    bf16_t* __restrict__ outs)           // [5][2016][256]
{
  __shared__ bf16_t h_sh[BB * HH];
  const int s    = blockIdx.x;          // stripe 0..125 (j index; start = s+1)
  const int tid  = threadIdx.x;
  const int lane = tid & 31;
  const int w    = tid >> 5;
  const int jb   = w * 16;
  const int n    = lane & 15;
  const int hi   = (lane >> 4) & 1;

  // h0 = seg_start[s] stripe; outs[0] = h0
  for (int i = tid; i < BB * HH; i += 512) {
    bf16_t v = seg[(size_t)s * BB * HH + i];
    h_sh[i] = v;
    outs[(size_t)(s * BB) * HH + i] = v;
  }
  __syncthreads();

  float bias_r[4];
#pragma unroll
  for (int g = 0; g < 4; ++g) bias_r[g] = bias[g * HH + jb + n];

  v8f c = {};
  for (int m = 0; m < MM; ++m) {
    int t_idx = s + 1 + m; if (t_idx > TT - 1) t_idx = TT - 1;
    v8f acc[4];
#pragma unroll
    for (int g = 0; g < 4; ++g)
#pragma unroll
      for (int r = 0; r < 8; ++r) acc[g][r] = bias_r[g];

    const bf16_t* xrow = embeds + (size_t)t_idx * BB * EE;
#pragma unroll 2
    for (int kk = 0; kk < 8; ++kk) {
      v16bf a = load_a_tile(xrow + kk * 32, EE, lane);
#pragma unroll
      for (int g = 0; g < 4; ++g) {
        v16bf b = load_b_tile(wih_t, EE, g * HH + jb, kk * 32, lane);
        acc[g] = wmma_bf16(a, b, acc[g]);
      }
    }
#pragma unroll 2
    for (int kk = 0; kk < 8; ++kk) {
      v16bf a = load_a_tile(h_sh + kk * 32, HH, lane);
#pragma unroll
      for (int g = 0; g < 4; ++g) {
        v16bf b = load_b_tile(whh_t, HH, g * HH + jb, kk * 32, lane);
        acc[g] = wmma_bf16(a, b, acc[g]);
      }
    }
    __syncthreads();
    float hv[8];
#pragma unroll
    for (int r = 0; r < 8; ++r) {
      float ig = sigf(acc[0][r]);
      float fg = sigf(acc[1][r]);
      float gg = tanhf(acc[2][r]);
      float og = sigf(acc[3][r]);
      c[r] = fg * c[r] + ig * gg;
      hv[r] = og * tanhf(c[r]);
    }
#pragma unroll
    for (int r = 0; r < 8; ++r) {
      int mr = r + 8 * hi;
      h_sh[mr * HH + jb + n] = (bf16_t)hv[r];
      outs[((size_t)(m + 1) * NN + s * BB + mr) * HH + jb + n] = (bf16_t)hv[r];
    }
    __syncthreads();
  }
}

// ---- Vocab projection with streaming logsumexp -----------------------------
// 315 blocks x 32-row (2-stripe) tiles; 16 waves sweep 500 column tiles of V.
// Each loaded B fragment feeds TWO WMMAs (one per row stripe), halving the
// per-block W_out traffic out of L2.
__global__ __launch_bounds__(512) void k_logits(
    const bf16_t* __restrict__ outs,    // [10080][256]
    const bf16_t* __restrict__ wout_t,  // [8000][256]
    const float*  __restrict__ b_out,   // [8000]
    const int*    __restrict__ tgt_col, // [10080]
    float* __restrict__ tgt_lp,         // [4*2016]
    float* __restrict__ eos_lp)         // [5*2016]
{
  __shared__ float red_m[16][32], red_s[16][32], red_e[16][32], red_t[16][32];
  const int tid  = threadIdx.x;
  const int lane = tid & 31;
  const int w    = tid >> 5;
  const int n    = lane & 15;
  const int hi   = (lane >> 4) & 1;
  const int qb   = blockIdx.x * 32;   // 32-row tile base

  // Preload A fragments for both stripes (rows reused across all 500 tiles)
  v16bf afrag0[8], afrag1[8];
#pragma unroll
  for (int kk = 0; kk < 8; ++kk) {
    afrag0[kk] = load_a_tile(outs + (size_t)qb * HH + kk * 32, HH, lane);
    afrag1[kk] = load_a_tile(outs + (size_t)(qb + 16) * HH + kk * 32, HH, lane);
  }

  int tc0[8], tc1[8];
#pragma unroll
  for (int r = 0; r < 8; ++r) {
    tc0[r] = tgt_col[qb + r + 8 * hi];
    tc1[r] = tgt_col[qb + 16 + r + 8 * hi];
  }

  float mx0[8], sm0[8], ev0[8], tv0[8];
  float mx1[8], sm1[8], ev1[8], tv1[8];
#pragma unroll
  for (int r = 0; r < 8; ++r) {
    mx0[r] = NEG_BIG; sm0[r] = 0.f; ev0[r] = NEG_BIG; tv0[r] = NEG_BIG;
    mx1[r] = NEG_BIG; sm1[r] = 0.f; ev1[r] = NEG_BIG; tv1[r] = NEG_BIG;
  }

  for (int ct = w; ct < VV / 16; ct += 16) {
    int c0 = ct * 16;
    // prefetch a future B tile's rows into cache
    if (c0 + 256 < VV) __builtin_prefetch(wout_t + (size_t)(c0 + 256) * HH, 0, 1);
    float bo = b_out[c0 + n];
    v8f acc0, acc1;
#pragma unroll
    for (int r = 0; r < 8; ++r) { acc0[r] = bo; acc1[r] = bo; }
#pragma unroll
    for (int kk = 0; kk < 8; ++kk) {
      v16bf b = load_b_tile(wout_t, HH, c0, kk * 32, lane);
      acc0 = wmma_bf16(afrag0[kk], b, acc0);
      acc1 = wmma_bf16(afrag1[kk], b, acc1);
    }
    int col = c0 + n;
#pragma unroll
    for (int r = 0; r < 8; ++r) {
      float v0 = acc0[r];
      float nm0 = fmaxf(mx0[r], v0);
      sm0[r] = sm0[r] * __expf(mx0[r] - nm0) + __expf(v0 - nm0);
      mx0[r] = nm0;
      if (col == 5)       ev0[r] = v0;   // EOS = 5
      if (col == tc0[r])  tv0[r] = v0;   // target token

      float v1 = acc1[r];
      float nm1 = fmaxf(mx1[r], v1);
      sm1[r] = sm1[r] * __expf(mx1[r] - nm1) + __expf(v1 - nm1);
      mx1[r] = nm1;
      if (col == 5)       ev1[r] = v1;
      if (col == tc1[r])  tv1[r] = v1;
    }
  }

  // Reduce across the 16 lanes that share rows (within each half)
#pragma unroll
  for (int r = 0; r < 8; ++r) {
    float m1 = mx0[r], s1 = sm0[r], e1 = ev0[r], t1 = tv0[r];
    float m3 = mx1[r], s3 = sm1[r], e3 = ev1[r], t3 = tv1[r];
    for (int off = 1; off < 16; off <<= 1) {
      float m2 = __shfl_xor(m1, off, 16);
      float s2 = __shfl_xor(s1, off, 16);
      float nm = fmaxf(m1, m2);
      s1 = s1 * __expf(m1 - nm) + s2 * __expf(m2 - nm);
      m1 = nm;
      e1 = fmaxf(e1, __shfl_xor(e1, off, 16));
      t1 = fmaxf(t1, __shfl_xor(t1, off, 16));

      float m4 = __shfl_xor(m3, off, 16);
      float s4 = __shfl_xor(s3, off, 16);
      float nm2 = fmaxf(m3, m4);
      s3 = s3 * __expf(m3 - nm2) + s4 * __expf(m4 - nm2);
      m3 = nm2;
      e3 = fmaxf(e3, __shfl_xor(e3, off, 16));
      t3 = fmaxf(t3, __shfl_xor(t3, off, 16));
    }
    if (n == 0) {
      int row = r + 8 * hi;
      red_m[w][row] = m1;      red_s[w][row] = s1;
      red_e[w][row] = e1;      red_t[w][row] = t1;
      red_m[w][row + 16] = m3; red_s[w][row + 16] = s3;
      red_e[w][row + 16] = e3; red_t[w][row + 16] = t3;
    }
  }
  __syncthreads();

  if (tid < 32) {
    int row = tid;
    float m1 = red_m[0][row], s1 = red_s[0][row];
    float e1 = red_e[0][row], t1 = red_t[0][row];
    for (int ww = 1; ww < 16; ++ww) {
      float m2 = red_m[ww][row], s2 = red_s[ww][row];
      float nm = fmaxf(m1, m2);
      s1 = s1 * __expf(m1 - nm) + s2 * __expf(m2 - nm);
      m1 = nm;
      e1 = fmaxf(e1, red_e[ww][row]);
      t1 = fmaxf(t1, red_t[ww][row]);
    }
    float logZ = m1 + __logf(s1);
    int q = qb + row;
    eos_lp[q] = e1 - logZ;
    if (q < MM * NN) tgt_lp[q] = t1 - logZ;
  }
}

// ---- Final DP: logpy, alpha scan, nll ---------------------------------------
__global__ __launch_bounds__(256) void k_dp(
    const float* __restrict__ tgt_lp,  // [4][2016]
    const float* __restrict__ eos_lp,  // [5][2016]
    const int*   __restrict__ x,       // [16][128]
    const int*   __restrict__ lengths, // [16]
    float* __restrict__ out)
{
  __shared__ float logpy[TT - 1][MM][BB];  // [127][4][16]
  __shared__ float alpha[TT - 1][BB];      // [127][16]
  __shared__ float red[BB];
  const int tid = threadIdx.x;

  for (int idx = tid; idx < (TT - 1) * MM * BB; idx += 256) {
    int b = idx & 15;
    int m = (idx >> 4) & 3;
    int t = idx >> 6;
    float v;
    if (t == 0) {
      v = (m == 0) ? 0.f : -LOGINF_F;
    } else {
      int j = t - 1;            // 0..125
      int nr = j * BB + b;
      int L = m + 1;            // segment length
      if (L <= (TT - 2) - j) {  // valid: L <= 126 - j
        float cum = 0.f;
        for (int mm2 = 0; mm2 <= m; ++mm2) cum += tgt_lp[mm2 * NN + nr];
        float cs = 0.f, s0v = 0.f;
        for (int mm2 = 0; mm2 <= m; ++mm2) {
          int tt = j + 1 + mm2; if (tt > TT - 1) tt = TT - 1;
          int tok = x[b * TT + tt];
          float sg = (tok == 2 || tok == 3 || tok == 7) ? -LOGINF_F : 0.f;
          if (mm2 == 0) { if (L >= 2) s0v = sg; }
          else cs += sg;
        }
        v = cum + cs + s0v + eos_lp[(m + 1) * NN + nr];
      } else {
        v = -LOGINF_F;
      }
    }
    logpy[t][m][b] = v;
  }
  for (int idx = tid; idx < (TT - 1) * BB; idx += 256)
    alpha[idx >> 4][idx & 15] = (idx < BB) ? 0.f : -LOGINF_F;
  __syncthreads();

  if (tid < BB) {
    int b = tid;
    for (int t = 1; t < TT - 1; ++t) {
      float terms[4];
      float mxv = NEG_BIG;
      for (int k = 1; k <= MM; ++k) {
        int jr = t - k;
        float tm = (jr >= 0) ? (alpha[jr][b] + logpy[jr + 1][k - 1][b]) : -LOGINF_F;
        terms[k - 1] = tm;
        mxv = fmaxf(mxv, tm);
      }
      float s = 0.f;
      for (int k = 0; k < MM; ++k) s += __expf(terms[k] - mxv);
      alpha[t][b] = mxv + __logf(s);
    }
    red[b] = -alpha[lengths[b] - 2][b];
  }
  __syncthreads();
  if (tid == 0) {
    float nll = 0.f; int tot = 0;
    for (int b = 0; b < BB; ++b) { nll += red[b]; tot += lengths[b]; }
    out[0] = nll / (float)(tot - 2 * BB);
  }
}

// ---------------------------------------------------------------------------
extern "C" void kernel_launch(void* const* d_in, const int* in_sizes, int n_in,
                              void* d_out, int out_size, void* d_ws, size_t ws_size,
                              hipStream_t stream) {
  const int*   x        = (const int*)  d_in[0];
  const int*   lengths  = (const int*)  d_in[1];
  const float* emb      = (const float*)d_in[2];
  const float* enc_Wih  = (const float*)d_in[3];
  const float* enc_Whh  = (const float*)d_in[4];
  const float* enc_b    = (const float*)d_in[5];
  const float* W_start  = (const float*)d_in[6];
  const float* b_start  = (const float*)d_in[7];
  const float* dec_Wih  = (const float*)d_in[8];
  const float* dec_Whh  = (const float*)d_in[9];
  const float* dec_b    = (const float*)d_in[10];
  const float* W_out    = (const float*)d_in[11];
  const float* b_out    = (const float*)d_in[12];
  float* out = (float*)d_out;

  // workspace carve-up (256B aligned)
  char* ws = (char*)d_ws;
  size_t off = 0;
  auto alloc = [&](size_t bytes) {
    char* p = ws + off;
    off += (bytes + 255) & ~(size_t)255;
    return p;
  };
  bf16_t* embeds_bf   = (bf16_t*)alloc((size_t)TT * BB * EE * 2);  // 1 MB
  bf16_t* encWih_t    = (bf16_t*)alloc((size_t)4 * HH * EE * 2);   // 512 KB
  bf16_t* encWhh_t    = (bf16_t*)alloc((size_t)4 * HH * HH * 2);
  bf16_t* decWih_t    = (bf16_t*)alloc((size_t)4 * HH * EE * 2);
  bf16_t* decWhh_t    = (bf16_t*)alloc((size_t)4 * HH * HH * 2);
  bf16_t* Wst_t       = (bf16_t*)alloc((size_t)HH * HH * 2);
  bf16_t* Wout_t      = (bf16_t*)alloc((size_t)VV * HH * 2);       // 4 MB
  bf16_t* enc_out_bf  = (bf16_t*)alloc((size_t)TT * BB * HH * 2);
  bf16_t* seg_bf      = (bf16_t*)alloc((size_t)TT * BB * HH * 2);
  bf16_t* outs_bf     = (bf16_t*)alloc((size_t)QQ * HH * 2);       // ~5 MB
  int*    tgt_col     = (int*)   alloc((size_t)QQ * 4);
  float*  tgt_lp      = (float*) alloc((size_t)MM * NN * 4);
  float*  eos_lp      = (float*) alloc((size_t)(MM + 1) * NN * 4);

  // --- prep: weight transposes (f32 -> bf16 [N][K]) ---
  auto tgrid = [](int K, int N) { return dim3((K * N + 255) / 256); };
  k_transpose_bf16<<<tgrid(EE, 4 * HH), 256, 0, stream>>>(enc_Wih, encWih_t, EE, 4 * HH);
  k_transpose_bf16<<<tgrid(HH, 4 * HH), 256, 0, stream>>>(enc_Whh, encWhh_t, HH, 4 * HH);
  k_transpose_bf16<<<tgrid(EE, 4 * HH), 256, 0, stream>>>(dec_Wih, decWih_t, EE, 4 * HH);
  k_transpose_bf16<<<tgrid(HH, 4 * HH), 256, 0, stream>>>(dec_Whh, decWhh_t, HH, 4 * HH);
  k_transpose_bf16<<<tgrid(HH, HH),     256, 0, stream>>>(W_start, Wst_t,    HH, HH);
  k_transpose_bf16<<<tgrid(HH, VV),     256, 0, stream>>>(W_out,   Wout_t,   HH, VV);

  k_embed<<<(TT * BB * EE + 255) / 256, 256, 0, stream>>>(x, emb, embeds_bf);
  k_tgtcol<<<(QQ + 255) / 256, 256, 0, stream>>>(x, tgt_col);

  // --- encoder LSTM (sequential; one WGP, 16 waves) ---
  k_enc_lstm<<<1, 512, 0, stream>>>(embeds_bf, encWih_t, encWhh_t, enc_b, enc_out_bf);

  // --- seg_start projection ---
  k_seg_start<<<256, 256, 0, stream>>>(enc_out_bf, Wst_t, b_start, seg_bf);

  // --- decoder LSTM over 126 independent stripes ---
  k_dec_lstm<<<JN, 512, 0, stream>>>(embeds_bf, seg_bf, decWih_t, decWhh_t, dec_b, outs_bf);

  // --- vocab projection + streaming logsumexp (2 row stripes per block) ---
  k_logits<<<QQ / 32, 512, 0, stream>>>(outs_bf, Wout_t, b_out, tgt_col, tgt_lp, eos_lp);

  // --- final DP -> scalar ---
  k_dp<<<1, 256, 0, stream>>>(tgt_lp, eos_lp, x, lengths, out);

  (void)in_sizes; (void)n_in; (void)out_size; (void)ws_size;
}
